// HQQLinear_75531294867847
// MI455X (gfx1250) — matrix operand order repaired and checked by reading, not compiled, not run on hardware
//
#include <hip/hip_runtime.h>

// Problem constants (from reference)
#define TOKENS 4096
#define IN_F   4096
#define OUT_F  11008
#define NGROUP (OUT_F / 64 * IN_F)   // 704512 groups
// o = r*172 + b ;  g = b*IN_F + i ; r = o/172

// GEMM tiling
#define BM 256
#define BN 128
#define BK 64
#define KPAD (BK + 8)   // pad LDS rows to 144B to break bank conflicts

typedef __attribute__((ext_vector_type(16))) __bf16 v16bf;
typedef __attribute__((ext_vector_type(8)))  __bf16 v8bf;
typedef __attribute__((ext_vector_type(4)))  __bf16 v4bf;
typedef __attribute__((ext_vector_type(8)))  float  v8f;
typedef __attribute__((ext_vector_type(4)))  float  v4f;
typedef __attribute__((ext_vector_type(4)))  int    v4i;

__global__ __launch_bounds__(256)
void hqq_wmma_gemm(const float* __restrict__ x,
                   const int*   __restrict__ wq,
                   const float* __restrict__ scale,
                   const float* __restrict__ zero,
                   const float* __restrict__ bias,
                   float*       __restrict__ out)
{
    // double-buffered tiles: 2 * (256+128) * 72 * 2B = 108 KB (LDS/WGP = 320 KB)
    __shared__ __align__(16) __bf16 xsm[2][BM][KPAD];
    __shared__ __align__(16) __bf16 wsm[2][BN][KPAD];

    const int t     = threadIdx.x;
    const int lane  = t & 31;
    const int wave  = t >> 5;
    const int waveM = (wave & 3) << 6;   // 0,64,128,192
    const int waveN = (wave >> 2) << 6;  // 0,64

    const int m0 = blockIdx.x * BM;      // token block
    const int n0 = blockIdx.y * BN;      // out-feature block

    // --- dequant assignment: each thread owns one (n, 32-wide k chunk) ---
    const int nIdx  = t >> 1;            // 0..127 within tile
    const int kHalf = (t & 1) << 5;      // 0 or 32
    const int n     = n0 + nIdx;
    const int r     = n / 172;           // which of the 64 "rows" of W_r
    const int bgrp  = n - r * 172;
    const int shift = (r < 32) ? 4 : 0;  // hi nibble for r<32, lo nibble else
    const int*   wrow = wq    + (size_t)(r & 31) * NGROUP + (size_t)bgrp * IN_F;
    const float* srow = scale + (size_t)bgrp * IN_F;
    const float* zrow = zero  + (size_t)bgrp * IN_F;

    // stage one BK-tile (x: f32->bf16, W: 4-bit dequant->bf16) into buffer `buf`
    auto stage = [&](int k0, int buf) {
        #pragma unroll
        for (int i = 0; i < 16; ++i) {
            const int c   = t + (i << 8);
            const int row = c >> 4;
            const int col = (c & 15) << 2;
            const float4 v = *(const float4*)(x + (size_t)(m0 + row) * IN_F + k0 + col);
            v4bf h;
            h[0] = (__bf16)v.x; h[1] = (__bf16)v.y;
            h[2] = (__bf16)v.z; h[3] = (__bf16)v.w;
            *(v4bf*)&xsm[buf][row][col] = h;
        }
        const int kb = k0 + kHalf;
        #pragma unroll
        for (int c8 = 0; c8 < 8; ++c8) {
            const v4i q4 = *(const v4i*)(wrow + kb + (c8 << 2));
            const v4f s4 = *(const v4f*)(srow + kb + (c8 << 2));
            const v4f z4 = *(const v4f*)(zrow + kb + (c8 << 2));
            v4bf o4;
            #pragma unroll
            for (int e = 0; e < 4; ++e) {
                const float q = (float)((q4[e] >> shift) & 0xF);
                o4[e] = (__bf16)((q - z4[e]) * s4[e]);
            }
            *(v4bf*)&wsm[buf][nIdx][kHalf + (c8 << 2)] = o4;
        }
    };

    v8f acc[4][4];
    const v8f z8 = {};
    #pragma unroll
    for (int i = 0; i < 4; ++i)
        #pragma unroll
        for (int j = 0; j < 4; ++j) acc[i][j] = z8;

    // prologue: stage tile 0
    stage(0, 0);
    __syncthreads();

    int cur = 0;
    for (int k0 = 0; k0 < IN_F; k0 += BK) {
        // stage next tile into the other buffer; scheduler overlaps these
        // global loads + dequant VALU with the WMMA stream below
        if (k0 + BK < IN_F) stage(k0 + BK, cur ^ 1);

        // ---- compute: 2 k-steps of 32, 4x4 WMMA tiles per wave ----
        #pragma unroll
        for (int kk = 0; kk < BK; kk += 32) {
            v16bf afrag[4], bfrag[4];
            // A fragment: lane<16 -> K 0..7 & 16..23 ; lane>=16 -> K 8..15 & 24..31
            const int arow  = waveM + (lane & 15);
            const int akoff = kk + ((lane >> 4) << 3);
            #pragma unroll
            for (int mi = 0; mi < 4; ++mi) {
                const v8bf lo = *(const v8bf*)&xsm[cur][arow + (mi << 4)][akoff];
                const v8bf hi = *(const v8bf*)&xsm[cur][arow + (mi << 4)][akoff + 16];
                afrag[mi] = __builtin_shufflevector(lo, hi,
                    0,1,2,3,4,5,6,7,8,9,10,11,12,13,14,15);
            }
            // B fragment: lane<16 -> K 0..7 & 16..23 of column N=lane ; lane>=16 -> K 8..15 & 24..31
            const int bcol  = waveN + (lane & 15);
            const int bkoff = kk + ((lane >> 4) << 3);
            #pragma unroll
            for (int ni = 0; ni < 4; ++ni) {
                const v8bf lo = *(const v8bf*)&wsm[cur][bcol + (ni << 4)][bkoff];
                const v8bf hi = *(const v8bf*)&wsm[cur][bcol + (ni << 4)][bkoff + 16];
                bfrag[ni] = __builtin_shufflevector(lo, hi,
                    0,1,2,3,4,5,6,7,8,9,10,11,12,13,14,15);
            }
            #pragma unroll
            for (int mi = 0; mi < 4; ++mi)
                #pragma unroll
                for (int ni = 0; ni < 4; ++ni)
                    acc[mi][ni] = __builtin_amdgcn_wmma_f32_16x16x32_bf16(
                        false, afrag[mi], false, bfrag[ni],
                        (short)0, acc[mi][ni], false, false);
        }
        __syncthreads();   // readers of buf[cur] done; writers of buf[cur^1] done
        cur ^= 1;
    }

    // ---- epilogue: C layout is VGPR j -> M row, lane -> N col; add bias ----
    // Output (180 MB) is write-once: non-temporal stores keep the L2-resident
    // x / W_q / scale / zero working set (~160 MB of 192 MB L2) from evicting.
    const int crow0 = m0 + waveM + ((lane >> 4) << 3);
    const int ccol0 = n0 + waveN + (lane & 15);
    #pragma unroll
    for (int ni = 0; ni < 4; ++ni) {
        const int   o  = ccol0 + (ni << 4);
        const float bv = bias[o];
        #pragma unroll
        for (int mi = 0; mi < 4; ++mi) {
            const size_t rbase = (size_t)(crow0 + (mi << 4)) * OUT_F + o;
            #pragma unroll
            for (int j = 0; j < 8; ++j)
                __builtin_nontemporal_store(acc[mi][ni][j] + bv,
                                            &out[rbase + (size_t)j * OUT_F]);
        }
    }
}

extern "C" void kernel_launch(void* const* d_in, const int* in_sizes, int n_in,
                              void* d_out, int out_size, void* d_ws, size_t ws_size,
                              hipStream_t stream) {
    (void)in_sizes; (void)n_in; (void)out_size; (void)d_ws; (void)ws_size;
    const float* x     = (const float*)d_in[0];
    const int*   wq    = (const int*)  d_in[1];
    const float* scale = (const float*)d_in[2];
    const float* zero  = (const float*)d_in[3];
    const float* bias  = (const float*)d_in[4];
    float*       out   = (float*)d_out;

    dim3 grid(TOKENS / BM, OUT_F / BN);   // 16 x 86 blocks
    hqq_wmma_gemm<<<grid, 256, 0, stream>>>(x, wq, scale, zero, bias, out);
}
